// SCNN_11897059410768
// MI455X (gfx1250) — compile-verified, matrix operand order checked by reading, not run on hardware
//
#include <hip/hip_runtime.h>

typedef __attribute__((ext_vector_type(16))) __bf16 v16bf;
typedef __attribute__((ext_vector_type(8)))  float  v8f;
typedef __attribute__((ext_vector_type(4), aligned(4))) float f32x4u; // 4B-aligned vec load

__device__ __forceinline__ unsigned short f2bf(float f) {
    unsigned u = __float_as_uint(f);
    unsigned r = u + 0x7FFFu + ((u >> 16) & 1u);   // round-to-nearest-even
    return (unsigned short)(r >> 16);
}

// ---------------------------------------------------------------------------
// Fold BN (+ optional conv bias) into per-channel scale/bias.
// ---------------------------------------------------------------------------
__global__ void fold_bn_k(const float* __restrict__ g, const float* __restrict__ b,
                          const float* __restrict__ m, const float* __restrict__ v,
                          const float* __restrict__ cb,
                          float* __restrict__ scale, float* __restrict__ bias, int C) {
    int i = blockIdx.x * blockDim.x + threadIdx.x;
    if (i >= C) return;
    float s = g[i] * rsqrtf(v[i] + 1e-5f);
    scale[i] = s;
    float c = cb ? cb[i] : 0.f;
    bias[i] = (c - m[i]) * s + b[i];
}

// ---------------------------------------------------------------------------
// Implicit-GEMM conv (stride 1, pad=(ksz/2)*dil), NCHW f32 in/out.
// M=C_out, N=H*W, K=C_in*ksz^2.  bf16 WMMA, f32 accumulate.
// Block: 256 thr = 8 waves; block tile 32(M) x 128(N); wave tile 16x32
// (two accumulators share one A fragment -> 2 v_wmma per K-step).
// Fast path (C_in%32==0): tap-major K order, so each 32-K chunk is one filter
// tap -> B staging uses contiguous 4-float row loads (global_load_b128).
// ---------------------------------------------------------------------------
__global__ __launch_bounds__(256) void conv_wmma_k(
    const float* __restrict__ x, const float* __restrict__ wgt,
    const float* __restrict__ scale, const float* __restrict__ bias,
    float* __restrict__ y,
    int C_in, int C_out, int H, int W, int dil, int ksz, int relu) {

    const int HW = H * W;
    const int KK = ksz * ksz;
    const int K  = C_in * KK;
    const int n  = blockIdx.z;
    const int m0 = blockIdx.y * 32;
    const int p0 = blockIdx.x * 128;

    __shared__ unsigned short As[2 * 512] __attribute__((aligned(32)));
    __shared__ unsigned short Bs[8 * 512] __attribute__((aligned(32)));

    const int tid  = threadIdx.x;
    const int lane = tid & 31;
    const int wv   = tid >> 5;
    const int msub = wv & 1;    // 0..1  (M subtile)
    const int nq   = wv >> 1;   // 0..3  (N pair: positions nq*32 .. nq*32+31)

    v8f acc0 = {}, acc1 = {};
    const float* xn = x + (size_t)n * C_in * HW;
    const int half = ksz >> 1;
    const bool fast = (C_in & 31) == 0;

    for (int k0 = 0; k0 < K; k0 += 32) {
        // chunk-constant tap for fast (tap-major) path
        int t_f = 0, cinb = 0, r_f = 0, s_f = 0;
        if (fast) {
            t_f  = k0 / C_in;
            cinb = k0 - t_f * C_in;
            r_f  = t_f / ksz;
            s_f  = t_f - r_f * ksz;
        }
        __syncthreads();
        if (k0 + 32 < K)
            __builtin_prefetch(&wgt[(size_t)m0 * K + k0 + 32], 0, 0);

        // ---- stage A (weights): 2 subtiles x 16 rows x 32 k
        for (int e = tid; e < 1024; e += 256) {
            int sub = e >> 9, r = (e >> 5) & 15, kk = e & 31;
            int m = m0 + sub * 16 + r;
            float v = 0.f;
            if (m < C_out) {
                if (fast) {
                    v = wgt[((size_t)m * C_in + (cinb + kk)) * KK + t_f];
                } else {
                    int k = k0 + kk;
                    if (k < K) v = wgt[(size_t)m * K + k];
                }
            }
            int l = r + ((kk & 8) ? 16 : 0);
            int j = ((kk & 16) ? 8 : 0) + (kk & 7);
            As[sub * 512 + l * 16 + j] = f2bf(v);
        }

        // ---- stage B (im2col): 32 k x 128 positions
        if (fast) {
            // quad-vectorized: 1024 quads, 4 per thread
            for (int u = tid; u < 1024; u += 256) {
                int kk = u >> 5;
                int np = (u & 31) << 2;          // 0,4,...,124
                int p  = p0 + np;
                int cin = cinb + kk;
                float v4[4] = {0.f, 0.f, 0.f, 0.f};
                int hh = p / W, ww = p - hh * W;
                int ih = hh + (r_f - half) * dil;
                int iwb = ww + (s_f - half) * dil;
                if (p + 3 < HW && ww + 3 < W &&
                    ih >= 0 && ih < H && iwb >= 0 && iwb + 3 < W) {
                    f32x4u q = *(const f32x4u*)(xn + (size_t)cin * HW + ih * W + iwb);
                    v4[0] = q.x; v4[1] = q.y; v4[2] = q.z; v4[3] = q.w;
                } else {
#pragma unroll
                    for (int i = 0; i < 4; ++i) {
                        int pi = p + i;
                        if (pi < HW) {
                            int h2 = pi / W, w2 = pi - h2 * W;
                            int ih2 = h2 + (r_f - half) * dil;
                            int iw2 = w2 + (s_f - half) * dil;
                            if (ih2 >= 0 && ih2 < H && iw2 >= 0 && iw2 < W)
                                v4[i] = xn[(size_t)cin * HW + ih2 * W + iw2];
                        }
                    }
                }
                int sub = np >> 4;               // quad never crosses a 16-group
                int c   = np & 15;
                int j   = ((kk & 16) ? 8 : 0) + (kk & 7);
                int lhi = (kk & 8) ? 16 : 0;
#pragma unroll
                for (int i = 0; i < 4; ++i)
                    Bs[sub * 512 + (c + i + lhi) * 16 + j] = f2bf(v4[i]);
            }
        } else {
            // generic scalar gather (layer0: C_in=3, K=27)
            for (int e = tid; e < 4096; e += 256) {
                int sub = e >> 9, c = (e >> 5) & 15, kk = e & 31;
                int p = p0 + sub * 16 + c, k = k0 + kk;
                float v = 0.f;
                if (p < HW && k < K) {
                    int cin = k / KK, t = k - cin * KK;
                    int hh = p / W, ww = p - hh * W;
                    int r = t / ksz, s = t - r * ksz;
                    int ih = hh + (r - half) * dil;
                    int iw = ww + (s - half) * dil;
                    if (ih >= 0 && ih < H && iw >= 0 && iw < W)
                        v = xn[(size_t)cin * HW + ih * W + iw];
                }
                int l = c + ((kk & 8) ? 16 : 0);
                int j = ((kk & 16) ? 8 : 0) + (kk & 7);
                Bs[sub * 512 + l * 16 + j] = f2bf(v);
            }
        }
        __syncthreads();

        v16bf a  = *(const v16bf*)&As[msub * 512 + lane * 16];
        v16bf b0 = *(const v16bf*)&Bs[(nq * 2 + 0) * 512 + lane * 16];
        v16bf b1 = *(const v16bf*)&Bs[(nq * 2 + 1) * 512 + lane * 16];
        acc0 = __builtin_amdgcn_wmma_f32_16x16x32_bf16(false, a, false, b0,
                                                       (short)0, acc0, false, false);
        acc1 = __builtin_amdgcn_wmma_f32_16x16x32_bf16(false, a, false, b1,
                                                       (short)0, acc1, false, false);
    }

    // ---- epilogue (C/D layout: N = lane&15, M = v + 8*(lane>=16))
    int mb = m0 + msub * 16 + ((lane >> 4) ? 8 : 0);
    v8f accs[2] = {acc0, acc1};
#pragma unroll
    for (int pair = 0; pair < 2; ++pair) {
        int p = p0 + nq * 32 + pair * 16 + (lane & 15);
        if (p < HW) {
#pragma unroll
            for (int v = 0; v < 8; ++v) {
                int m = mb + v;
                if (m < C_out) {
                    float s  = scale ? scale[m] : 1.f;
                    float bb = bias  ? bias[m]  : 0.f;
                    float val = accs[pair][v] * s + bb;
                    if (relu) val = fmaxf(val, 0.f);
                    y[(size_t)n * C_out * HW + (size_t)m * HW + p] = val;
                }
            }
        }
    }
}

// ---------------------------------------------------------------------------
// One serial message-passing step (in-place): line li += relu(conv1d(line lp)) (vertical; horizontal reads line li)
// GEMM: M=C(=128), N=B*S, K=C*9.
// ---------------------------------------------------------------------------
__global__ __launch_bounds__(256) void mp_step_k(
    float* __restrict__ xb, const float* __restrict__ wgt,
    int B, int C, int H, int W, int vertical, int li, int lp) {

    const int S = vertical ? W : H;
    const int N = B * S;
    const int K = C * 9;
    const int m0 = blockIdx.y * 32;
    const int q0 = blockIdx.x * 64;

    __shared__ unsigned short As[2 * 512] __attribute__((aligned(32)));
    __shared__ unsigned short Bs[4 * 512] __attribute__((aligned(32)));

    const int tid  = threadIdx.x;
    const int lane = tid & 31;
    const int wv   = tid >> 5;
    const int msub = wv & 1;
    const int nsub = wv >> 1;

    v8f acc = {};

    for (int k0 = 0; k0 < K; k0 += 32) {
        __syncthreads();
        for (int e = tid; e < 1024; e += 256) {
            int sub = e >> 9, r = (e >> 5) & 15, kk = e & 31;
            int m = m0 + sub * 16 + r, k = k0 + kk;
            float v = 0.f;
            if (m < C && k < K) v = wgt[(size_t)m * K + k];
            int l = r + ((kk & 8) ? 16 : 0);
            int j = ((kk & 16) ? 8 : 0) + (kk & 7);
            As[sub * 512 + l * 16 + j] = f2bf(v);
        }
        for (int e = tid; e < 2048; e += 256) {
            int sub = e >> 9, c = (e >> 5) & 15, kk = e & 31;
            int q = q0 + sub * 16 + c, k = k0 + kk;
            float v = 0.f;
            if (q < N && k < K) {
                int cin = k / 9, t = k - cin * 9;
                int n = q / S, s = q - n * S;
                int sp = s + t - 4;
                if (sp >= 0 && sp < S) {
                    size_t off = vertical
                        ? (((size_t)n * C + cin) * H + lp) * W + sp
                        : (((size_t)n * C + cin) * H + sp) * W + li;
                    v = xb[off];
                }
            }
            int l = c + ((kk & 8) ? 16 : 0);
            int j = ((kk & 16) ? 8 : 0) + (kk & 7);
            Bs[sub * 512 + l * 16 + j] = f2bf(v);
        }
        __syncthreads();

        v16bf a = *(const v16bf*)&As[msub * 512 + lane * 16];
        v16bf b = *(const v16bf*)&Bs[nsub * 512 + lane * 16];
        acc = __builtin_amdgcn_wmma_f32_16x16x32_bf16(false, a, false, b,
                                                      (short)0, acc, false, false);
    }

    int q  = q0 + nsub * 16 + (lane & 15);
    int mb = m0 + msub * 16 + ((lane >> 4) ? 8 : 0);
    if (q < N) {
        int n = q / S, s = q - n * S;
#pragma unroll
        for (int v = 0; v < 8; ++v) {
            int m = mb + v;
            if (m < C) {
                size_t off = vertical
                    ? (((size_t)n * C + m) * H + li) * W + s
                    : (((size_t)n * C + m) * H + s) * W + li;
                xb[off] = xb[off] + fmaxf(acc[v], 0.f);
            }
        }
    }
}

// ---------------------------------------------------------------------------
__global__ void maxpool2_k(const float* __restrict__ in, float* __restrict__ out,
                           int NC, int H, int W) {
    int oh = H >> 1, ow = W >> 1;
    int i = blockIdx.x * blockDim.x + threadIdx.x;
    int total = NC * oh * ow;
    if (i >= total) return;
    int x = i % ow, y = (i / ow) % oh, nc = i / (ow * oh);
    const float* p = in + ((size_t)nc * H + 2 * y) * W + 2 * x;
    out[i] = fmaxf(fmaxf(p[0], p[1]), fmaxf(p[W], p[W + 1]));
}

__global__ void bilinear8_k(const float* __restrict__ in, float* __restrict__ out,
                            int NC, int H, int W, int OH, int OW) {
    int i = blockIdx.x * blockDim.x + threadIdx.x;
    int total = NC * OH * OW;
    if (i >= total) return;
    int ox = i % OW, oy = (i / OW) % OH, nc = i / (OW * OH);
    float fy = oy * (float)(H - 1) / (float)(OH - 1);
    float fx = ox * (float)(W - 1) / (float)(OW - 1);
    int y0 = (int)floorf(fy); int y1 = min(y0 + 1, H - 1); float wy = fy - y0;
    int x0 = (int)floorf(fx); int x1 = min(x0 + 1, W - 1); float wx = fx - x0;
    const float* p = in + (size_t)nc * H * W;
    float top = p[y0 * W + x0] * (1.f - wx) + p[y0 * W + x1] * wx;
    float bot = p[y1 * W + x0] * (1.f - wx) + p[y1 * W + x1] * wx;
    out[i] = top * (1.f - wy) + bot * wy;
}

__global__ void smax_pool_k(const float* __restrict__ seg, float* __restrict__ fcin) {
    int i = blockIdx.x * blockDim.x + threadIdx.x;
    if (i >= 4 * 18 * 50) return;
    int x = i % 50, y = (i / 50) % 18, n = i / 900;
    const float* p = seg + (size_t)n * 5 * 3600;
    float accv[5] = {0, 0, 0, 0, 0};
#pragma unroll
    for (int dy = 0; dy < 2; ++dy)
#pragma unroll
        for (int dx = 0; dx < 2; ++dx) {
            int h = 2 * y + dy, w = 2 * x + dx;
            float vals[5], mx = -1e30f;
#pragma unroll
            for (int c = 0; c < 5; ++c) { vals[c] = p[c * 3600 + h * 100 + w]; mx = fmaxf(mx, vals[c]); }
            float sum = 0.f;
#pragma unroll
            for (int c = 0; c < 5; ++c) { vals[c] = expf(vals[c] - mx); sum += vals[c]; }
            float inv = 1.f / sum;
#pragma unroll
            for (int c = 0; c < 5; ++c) accv[c] += vals[c] * inv;
        }
#pragma unroll
    for (int c = 0; c < 5; ++c)
        fcin[n * 4500 + c * 900 + y * 50 + x] = accv[c] * 0.25f;
}

__global__ void fc1_k(const float* __restrict__ a, const float* __restrict__ w,
                      const float* __restrict__ b, float* __restrict__ o) {
    int i = blockIdx.x * blockDim.x + threadIdx.x;
    if (i >= 4 * 128) return;
    int j = i & 127, n = i >> 7;
    const float* av = a + n * 4500;
    const float* wv = w + (size_t)j * 4500;
    float s = 0.f;
    for (int k = 0; k < 4500; ++k) s += av[k] * wv[k];
    o[i] = fmaxf(s + b[j], 0.f);
}

__global__ void fc2_k(const float* __restrict__ h, const float* __restrict__ w,
                      const float* __restrict__ b, float* __restrict__ o) {
    int i = threadIdx.x;
    if (i >= 16) return;
    int j = i & 3, n = i >> 2;
    float s = 0.f;
    for (int k = 0; k < 128; ++k) s += h[n * 128 + k] * w[j * 128 + k];
    o[i] = 1.f / (1.f + expf(-(s + b[j])));
}

__global__ void tail3_k(float* __restrict__ o) {
    if (threadIdx.x < 3) o[threadIdx.x] = 0.f;
}

// ---------------------------------------------------------------------------
extern "C" void kernel_launch(void* const* d_in, const int* in_sizes, int n_in,
                              void* d_out, int out_size, void* d_ws, size_t ws_size,
                              hipStream_t stream) {
    (void)in_sizes; (void)n_in; (void)out_size; (void)ws_size;
    const float* img = (const float*)d_in[0];
    const float *conv_w[13], *conv_b[13], *bn_g[13], *bn_b[13], *bn_m[13], *bn_v[13];
    for (int i = 0; i < 13; ++i) {
        conv_w[i] = (const float*)d_in[1 + i];
        conv_b[i] = (const float*)d_in[14 + i];
        bn_g[i]   = (const float*)d_in[27 + i];
        bn_b[i]   = (const float*)d_in[40 + i];
        bn_m[i]   = (const float*)d_in[53 + i];
        bn_v[i]   = (const float*)d_in[66 + i];
    }
    const float* l1_w1 = (const float*)d_in[79];
    const float* l1_g  = (const float*)d_in[80];
    const float* l1_bb = (const float*)d_in[81];
    const float* l1_m  = (const float*)d_in[82];
    const float* l1_v  = (const float*)d_in[83];
    const float* l1_w2 = (const float*)d_in[84];
    const float* mp_ud = (const float*)d_in[85];
    const float* mp_du = (const float*)d_in[86];
    const float* mp_lr = (const float*)d_in[87];
    const float* mp_rl = (const float*)d_in[88];
    const float* l2_w  = (const float*)d_in[89];
    const float* l2_b  = (const float*)d_in[90];
    const float* fc1_w = (const float*)d_in[91];
    const float* fc1_b = (const float*)d_in[92];
    const float* fc2_w = (const float*)d_in[93];
    const float* fc2_b = (const float*)d_in[94];

    float* out = (float*)d_out;
    float* ws  = (float*)d_ws;

    const size_t BUF = 58982400;       // 4*64*288*800 (largest activation)
    float* bufs[2] = { ws, ws + BUF };
    float* sb   = ws + 2 * BUF;        // 2048: scale[1024] + bias[1024]
    float* fcin = sb + 2048;           // 4*4500
    float* fch  = fcin + 18000;        // 4*128
    float* seg  = fch + 512;           // 4*5*3600

    const int OC[13]  = {64,64,128,128,256,256,256,512,512,512,512,512,512};
    const int DIL[13] = {1,1,1,1,1,1,1,1,1,1,2,2,2};
    const bool PL[13] = {false,true,false,true,false,false,true,
                         false,false,false,false,false,false};

    int H = 288, W = 800, C = 3, bi = 0;
    const float* src = img;

    // ---- VGG backbone
    for (int i = 0; i < 13; ++i) {
        fold_bn_k<<<dim3((OC[i] + 255) / 256), dim3(256), 0, stream>>>(
            bn_g[i], bn_b[i], bn_m[i], bn_v[i], conv_b[i], sb, sb + 1024, OC[i]);
        dim3 g((H * W + 127) / 128, (OC[i] + 31) / 32, 4);
        conv_wmma_k<<<g, dim3(256), 0, stream>>>(
            src, conv_w[i], sb, sb + 1024, bufs[bi], C, OC[i], H, W, DIL[i], 3, 1);
        src = bufs[bi]; bi ^= 1; C = OC[i];
        if (PL[i]) {
            int total = 4 * C * (H / 2) * (W / 2);
            maxpool2_k<<<dim3((total + 255) / 256), dim3(256), 0, stream>>>(
                src, bufs[bi], 4 * C, H, W);
            src = bufs[bi]; bi ^= 1; H /= 2; W /= 2;
        }
    }
    // H=36, W=100, C=512

    // ---- layer1
    fold_bn_k<<<dim3(4), dim3(256), 0, stream>>>(
        l1_g, l1_bb, l1_m, l1_v, nullptr, sb, sb + 1024, 1024);
    {
        dim3 g((H * W + 127) / 128, 32, 4);
        conv_wmma_k<<<g, dim3(256), 0, stream>>>(
            src, l1_w1, sb, sb + 1024, bufs[bi], 512, 1024, H, W, 4, 3, 1);
        src = bufs[bi]; bi ^= 1;
    }
    {
        dim3 g((H * W + 127) / 128, 4, 4);
        conv_wmma_k<<<g, dim3(256), 0, stream>>>(
            src, l1_w2, nullptr, nullptr, bufs[bi], 1024, 128, H, W, 1, 1, 1);
        src = bufs[bi]; bi ^= 1;
    }
    float* mp = (float*)src;   // 4x128x36x100, updated in place

    // ---- message passing (serial recurrences; stream order = dependency)
    dim3 gv((4 * W + 63) / 64, 4, 1);
    dim3 gh((4 * H + 63) / 64, 4, 1);
    for (int i = 1; i < H; ++i)
        mp_step_k<<<gv, dim3(256), 0, stream>>>(mp, mp_ud, 4, 128, H, W, 1, i, i - 1);
    for (int i = H - 2; i >= 0; --i)
        mp_step_k<<<gv, dim3(256), 0, stream>>>(mp, mp_du, 4, 128, H, W, 1, i, i + 1);
    for (int i = 1; i < W; ++i)
        mp_step_k<<<gh, dim3(256), 0, stream>>>(mp, mp_lr, 4, 128, H, W, 0, i, i - 1);
    for (int i = W - 2; i >= 0; --i)
        mp_step_k<<<gh, dim3(256), 0, stream>>>(mp, mp_rl, 4, 128, H, W, 0, i, i + 1);

    // ---- layer2: 1x1 128->5 (+bias)
    {
        dim3 g((H * W + 127) / 128, 1, 4);
        conv_wmma_k<<<g, dim3(256), 0, stream>>>(
            mp, l2_w, nullptr, l2_b, seg, 128, 5, H, W, 1, 1, 0);
    }

    // ---- seg_pred: bilinear x8 (align_corners)
    {
        int total = 4 * 5 * 288 * 800;
        bilinear8_k<<<dim3((total + 255) / 256), dim3(256), 0, stream>>>(
            seg, out, 20, H, W, 288, 800);
    }

    // ---- layer3 head
    smax_pool_k<<<dim3((3600 + 255) / 256), dim3(256), 0, stream>>>(seg, fcin);
    fc1_k<<<dim3(2), dim3(256), 0, stream>>>(fcin, fc1_w, fc1_b, fch);
    fc2_k<<<dim3(1), dim3(32), 0, stream>>>(fch, fc2_w, fc2_b, out + 4608000);
    tail3_k<<<dim3(1), dim3(32), 0, stream>>>(out + 4608016);
}
